// HopfieldHNL_71279277245075
// MI455X (gfx1250) — compile-verified, matrix operand order checked by reading, not run on hardware
//
#include <hip/hip_runtime.h>
#include <hip/hip_bf16.h>

// Sizes from the reference
constexpr int N    = 1024;
constexpr int IN   = 512;
constexpr int H    = 8;
constexpr int D    = 64;
constexpr int M    = 16384;
constexpr int Bb   = 2048;   // bins
constexpr int TOPK = 64;
constexpr int HD   = H * D;  // 512

typedef __attribute__((ext_vector_type(16))) __bf16 v16bf;
typedef __attribute__((ext_vector_type(8)))  float  v8f;
typedef int v4i_vs __attribute__((vector_size(16)));   // matches builtin param type

// ---- CDNA5 async global->LDS support (guarded; fallback = sync copy) ------
#if defined(__has_builtin)
#  if __has_builtin(__builtin_amdgcn_global_load_async_to_lds_b128)
#    define HAVE_ASYNC_LDS 1
#  endif
#endif
#ifndef HAVE_ASYNC_LDS
#  define HAVE_ASYNC_LDS 0
#endif

__device__ __forceinline__ void wait_async0() {
#if defined(__has_builtin) && __has_builtin(__builtin_amdgcn_s_wait_asynccnt)
    __builtin_amdgcn_s_wait_asynccnt(0);
#else
    asm volatile("s_wait_asynccnt 0" ::: "memory");
#endif
}

// 16-byte async copy global->LDS (GLOBAL_LOAD_ASYNC_TO_LDS_B128, ASYNCcnt).
// Integer round-trip builds the addrspace pointers: global generic == as(1);
// low 32 bits of a generic shared pointer are the LDS byte offset.
__device__ __forceinline__ void async_cp16(void* lds, const void* g) {
#if HAVE_ASYNC_LDS
    auto gp = (__attribute__((address_space(1))) v4i_vs*)(unsigned long long)(uintptr_t)g;
    auto lp = (__attribute__((address_space(3))) v4i_vs*)(unsigned int)(uintptr_t)lds;
    __builtin_amdgcn_global_load_async_to_lds_b128(gp, lp, 0, 0);
#endif
}

// A-matrix 16x32 bf16 VGPR layout (ISA 7.12.2): element i of the v16bf frag
// lives in VGPR i/2, half i%2; K index depends on lane-half `hi`.
__device__ __forceinline__ int kmap(int i, int hi) {
    return (((i >> 1) & 3) << 1) + (i & 1) + ((i >> 3) << 4) + (hi << 3);
}

// ---------------------------------------------------------------------------
// Stage 0a: bin_proj f32 -> bf16
__global__ void k_cvt(const float* __restrict__ in, __bf16* __restrict__ out, int n) {
    int i = blockIdx.x * 256 + threadIdx.x;
    if (i < n) out[i] = (__bf16)in[i];
}

// Stage 0b: memories -> L2-normalized bf16 rows of length 64
__global__ void k_memnorm(const float* __restrict__ mem, __bf16* __restrict__ memn) {
    __shared__ float red[64];
    size_t base = (size_t)blockIdx.x * 64;
    int d = threadIdx.x;
    float v = mem[base + d];
    red[d] = v * v;
    __syncthreads();
    for (int s = 32; s > 0; s >>= 1) {
        if (d < s) red[d] += red[d + s];
        __syncthreads();
    }
    memn[base + d] = (__bf16)(v * rsqrtf(red[0]));
}

// ---------------------------------------------------------------------------
// Stage 1: q = x @ W_q^T  (bias added in k_qnorm).  bf16 WMMA, f32 accum.
__global__ void k_qproj(const float* __restrict__ x, const float* __restrict__ Wq,
                        float* __restrict__ q) {
    int wave = threadIdx.x >> 5, lane = threadIdx.x & 31;
    int tile = blockIdx.x * 8 + wave;        // 64 x 32 tiles
    int nt = tile >> 5, ct = tile & 31;
    int lr = lane & 15, hi = lane >> 4;
    int row = nt * 16 + lr;                  // n
    int col = ct * 16 + lr;                  // output channel
    v8f c = {0.f, 0.f, 0.f, 0.f, 0.f, 0.f, 0.f, 0.f};
    for (int kb = 0; kb < IN; kb += 32) {
        v16bf a, b;
#pragma unroll
        for (int i = 0; i < 16; ++i) {
            int k = kb + kmap(i, hi);
            a[i] = (__bf16)x[(size_t)row * IN + k];
            b[i] = (__bf16)Wq[(size_t)col * IN + k];
        }
        c = __builtin_amdgcn_wmma_f32_16x16x32_bf16(false, a, false, b,
                                                    (short)0, c, false, false);
    }
#pragma unroll
    for (int r = 0; r < 8; ++r) {
        int m = r + hi * 8;
        q[(size_t)(nt * 16 + m) * HD + ct * 16 + lr] = c[r];
    }
}

// Stage 1b: add bias, L2-normalize per (n,h) over D=64, emit bf16
__global__ void k_qnorm(const float* __restrict__ q, const float* __restrict__ bq,
                        __bf16* __restrict__ qn) {
    __shared__ float red[64];
    int n = blockIdx.x / H, h = blockIdx.x % H, d = threadIdx.x;
    float v = q[(size_t)n * HD + h * 64 + d] + bq[h * 64 + d];
    red[d] = v * v;
    __syncthreads();
    for (int s = 32; s > 0; s >>= 1) {
        if (d < s) red[d] += red[d + s];
        __syncthreads();
    }
    qn[((size_t)n * H + h) * 64 + d] = (__bf16)(v * rsqrtf(red[0]));
}

// ---------------------------------------------------------------------------
// Stage 2: bin_scores[n,b] = qn[n,h,:] . bin_proj[h,b,:]  (one head).
__global__ void k_scores(const __bf16* __restrict__ qn, const __bf16* __restrict__ bp,
                         float* __restrict__ sc, int h) {
    int wave = threadIdx.x >> 5, lane = threadIdx.x & 31;
    int tile = blockIdx.x * 8 + wave;        // 64 x 128 tiles
    int nt = tile >> 7, bt = tile & 127;
    int lr = lane & 15, hi = lane >> 4;
    int n = nt * 16 + lr, bcol = bt * 16 + lr;
    v8f c = {0.f, 0.f, 0.f, 0.f, 0.f, 0.f, 0.f, 0.f};
    for (int kb = 0; kb < D; kb += 32) {
        v16bf a, b;
#pragma unroll
        for (int i = 0; i < 16; ++i) {
            int k = kb + kmap(i, hi);
            a[i] = qn[((size_t)n * H + h) * 64 + k];
            b[i] = bp[((size_t)h * Bb + bcol) * 64 + k];
        }
        c = __builtin_amdgcn_wmma_f32_16x16x32_bf16(false, a, false, b,
                                                    (short)0, c, false, false);
    }
#pragma unroll
    for (int r = 0; r < 8; ++r)
        sc[(size_t)(nt * 16 + r + hi * 8) * Bb + bt * 16 + lr] = c[r];
}

// Stage 3: top-64 of 2048 per row via iterative masked max (one head).
__global__ void k_topk(const float* __restrict__ sc, int* __restrict__ idx, int h) {
    __shared__ float sv[Bb];
    __shared__ float rv[256];
    __shared__ int   ri[256];
    int n = blockIdx.x, tid = threadIdx.x;
    for (int j = tid; j < Bb; j += 256) sv[j] = sc[(size_t)n * Bb + j];
    __syncthreads();
    int* op = idx + ((size_t)n * H + h) * TOPK;
    for (int kk = 0; kk < TOPK; ++kk) {
        float bm = -3.4e38f; int bi = 0;
        for (int j = tid; j < Bb; j += 256) {
            float v = sv[j];
            if (v > bm) { bm = v; bi = j; }
        }
        rv[tid] = bm; ri[tid] = bi;
        __syncthreads();
        for (int s = 128; s > 0; s >>= 1) {
            if (tid < s && rv[tid + s] > rv[tid]) { rv[tid] = rv[tid + s]; ri[tid] = ri[tid + s]; }
            __syncthreads();
        }
        if (tid == 0) { op[kk] = ri[0]; sv[ri[0]] = -3.4e38f; }
        __syncthreads();
    }
}

// ---------------------------------------------------------------------------
// Stage 4 (dominant): attn[n, mt*16..+15] = sum_k W[h][m, idx[n,h,k]]
// W tile [16 x 2048] is one contiguous 128 KB block -> straight async
// global->LDS b128 copy (ASYNCcnt).  Row-major LDS is fine: gather lanes
// differ in random b, so banks spread as b % 64.
__global__ void k_attn(const float* __restrict__ W, const int* __restrict__ idx,
                       float* __restrict__ attnh, int h) {
    extern __shared__ float tile[];          // [16][2048] = 128 KB
    int tid = threadIdx.x, mt = blockIdx.x;
    const float* Wb = W + (size_t)h * M * Bb + (size_t)mt * 16 * Bb;
#if HAVE_ASYNC_LDS
    for (int i = tid; i < (16 * Bb) / 4; i += 256)
        async_cp16(&tile[i * 4], &Wb[i * 4]);
    wait_async0();
#else
    for (int i = tid; i < 16 * Bb; i += 256) tile[i] = Wb[i];
#endif
    __syncthreads();
    float acc[4][16];
#pragma unroll
    for (int nl = 0; nl < 4; ++nl)
#pragma unroll
        for (int m = 0; m < 16; ++m) acc[nl][m] = 0.f;
#pragma unroll
    for (int nl = 0; nl < 4; ++nl) {
        int n = nl * 256 + tid;
        const int* ip = idx + ((size_t)n * H + h) * TOPK;
        for (int k = 0; k < TOPK; ++k) {
            int b = ip[k];
#pragma unroll
            for (int m = 0; m < 16; ++m) acc[nl][m] += tile[m * Bb + b];
        }
    }
#pragma unroll
    for (int nl = 0; nl < 4; ++nl) {
        int n = nl * 256 + tid;
        float* op = attnh + (size_t)n * M + mt * 16;
#pragma unroll
        for (int m = 0; m < 16; ++m) op[m] = acc[nl][m];
    }
}

// Stage 5a: per row, max and sum of exp over transformed logits
// l = 10 * (2a/(1+a)),  a = raw/64
__global__ void k_maxsum(const float* __restrict__ attnh,
                         float* __restrict__ rmax, float* __restrict__ rsum) {
    __shared__ float red[256];
    int n = blockIdx.x, tid = threadIdx.x;
    const float* rp = attnh + (size_t)n * M;
    float lm = -3.4e38f;
    for (int j = tid; j < M; j += 256) {
        float a = rp[j] * (1.0f / TOPK);
        float l = 20.f * a / (1.f + a);
        lm = fmaxf(lm, l);
    }
    red[tid] = lm; __syncthreads();
    for (int s = 128; s > 0; s >>= 1) {
        if (tid < s) red[tid] = fmaxf(red[tid], red[tid + s]);
        __syncthreads();
    }
    float gm = red[0]; __syncthreads();
    float ls = 0.f;
    for (int j = tid; j < M; j += 256) {
        float a = rp[j] * (1.0f / TOPK);
        float l = 20.f * a / (1.f + a);
        ls += __expf(l - gm);
    }
    red[tid] = ls; __syncthreads();
    for (int s = 128; s > 0; s >>= 1) {
        if (tid < s) red[tid] += red[tid + s];
        __syncthreads();
    }
    if (tid == 0) { rmax[n] = gm; rsum[n] = red[0]; }
}

// Stage 5b: out[n, h*64+d] = sqrt(D) * sum_m probs[n,m] * memn[h,m,d]
// K=16384 reduction via bf16 WMMA; 128-wide m-chunks; mem chunk staged to
// LDS with async b128 copies; next attn chunk prefetched.
constexpr int CH = 128;
__global__ void k_out(const float* __restrict__ attnh, const __bf16* __restrict__ memn,
                      const float* __restrict__ rmax, const float* __restrict__ rsum,
                      float* __restrict__ out, int h) {
    __shared__ __bf16 pl[16 * CH];           // probs chunk [n=16][m=CH]  (4 KB)
    __shared__ __bf16 ml[CH * 64];           // mem chunk   [m=CH][d=64] (16 KB)
    int tid = threadIdx.x;
    int wave = tid >> 5, lane = tid & 31;
    int nt = blockIdx.x, dt = wave;
    int lr = lane & 15, hi = lane >> 4;
    v8f c = {0.f, 0.f, 0.f, 0.f, 0.f, 0.f, 0.f, 0.f};
    const __bf16* mb = memn + (size_t)h * M * 64;
    for (int m0 = 0; m0 < M; m0 += CH) {
        // mem chunk: contiguous CH*64 bf16 = 16 KB -> async copy
#if HAVE_ASYNC_LDS
        for (int i = tid; i < (CH * 64) / 8; i += 128)
            async_cp16(&ml[i * 8], &mb[(size_t)m0 * 64 + i * 8]);
#else
        for (int e = tid; e < CH * 64; e += 128)
            ml[e] = mb[(size_t)m0 * 64 + e];
#endif
        // probs chunk (computed, not copied)
        for (int e = tid; e < 16 * CH; e += 128) {
            int nr = e >> 7, mm = e & (CH - 1);
            int n = nt * 16 + nr;
            float a = attnh[(size_t)n * M + m0 + mm] * (1.0f / TOPK);
            float l = 20.f * a / (1.f + a);
            float p = __expf(l - rmax[n]) / rsum[n];
            pl[nr * CH + mm] = (__bf16)p;
            if (m0 + CH < M)
                __builtin_prefetch(&attnh[(size_t)n * M + m0 + CH + mm], 0, 1);
        }
#if HAVE_ASYNC_LDS
        wait_async0();
#endif
        __syncthreads();
#pragma unroll
        for (int kb = 0; kb < CH; kb += 32) {
            v16bf a, b;
#pragma unroll
            for (int i = 0; i < 16; ++i) {
                int k = kb + kmap(i, hi);
                a[i] = pl[lr * CH + k];
                b[i] = ml[k * 64 + dt * 16 + lr];
            }
            c = __builtin_amdgcn_wmma_f32_16x16x32_bf16(false, a, false, b,
                                                        (short)0, c, false, false);
        }
        __syncthreads();
    }
#pragma unroll
    for (int r = 0; r < 8; ++r) {
        int n = nt * 16 + r + hi * 8;
        out[(size_t)n * HD + h * 64 + dt * 16 + lr] = c[r] * 8.0f;  // sqrt(D)=8
    }
}

// ---------------------------------------------------------------------------
extern "C" void kernel_launch(void* const* d_in, const int* in_sizes, int n_in,
                              void* d_out, int out_size, void* d_ws, size_t ws_size,
                              hipStream_t stream) {
    (void)in_sizes; (void)n_in; (void)out_size; (void)ws_size;
    const float* x   = (const float*)d_in[0];
    const float* Wq  = (const float*)d_in[1];
    const float* bq  = (const float*)d_in[2];
    const float* bp  = (const float*)d_in[3];
    const float* Wm  = (const float*)d_in[4];
    const float* mem = (const float*)d_in[5];
    float* out = (float*)d_out;

    char* ws = (char*)d_ws;
    size_t off = 0;
    auto walloc = [&](size_t bytes) -> void* {
        void* p = ws + off;
        off += (bytes + 255) & ~(size_t)255;
        return p;
    };
    float*  q     = (float*)walloc((size_t)N * HD * 4);          // 2 MB
    __bf16* qn    = (__bf16*)walloc((size_t)N * HD * 2);         // 1 MB
    __bf16* bpn   = (__bf16*)walloc((size_t)H * Bb * D * 2);     // 2 MB
    __bf16* memn  = (__bf16*)walloc((size_t)H * M * D * 2);      // 16 MB
    float*  sc    = (float*)walloc((size_t)N * Bb * 4);          // 8 MB (per-head)
    int*    idx   = (int*)walloc((size_t)N * H * TOPK * 4);      // 2 MB
    float*  attnh = (float*)walloc((size_t)N * M * 4);           // 64 MB (per-head)
    float*  rmax  = (float*)walloc((size_t)N * 4);
    float*  rsum  = (float*)walloc((size_t)N * 4);

    const int attnLds = 16 * Bb * 4;  // 128 KB dynamic LDS (<= 320 KB/WGP)
    (void)hipFuncSetAttribute((const void*)k_attn,
                              hipFuncAttributeMaxDynamicSharedMemorySize, attnLds);

    k_cvt<<<(H * Bb * D + 255) / 256, 256, 0, stream>>>(bp, bpn, H * Bb * D);
    k_memnorm<<<H * M, 64, 0, stream>>>(mem, memn);
    k_qproj<<<256, 256, 0, stream>>>(x, Wq, q);
    k_qnorm<<<N * H, 64, 0, stream>>>(q, bq, qn);
    for (int h = 0; h < H; ++h) {
        k_scores<<<1024, 256, 0, stream>>>(qn, bpn, sc, h);
        k_topk<<<N, 256, 0, stream>>>(sc, idx, h);
    }
    for (int h = 0; h < H; ++h) {
        k_attn<<<M / 16, 256, attnLds, stream>>>(Wm, idx, attnh, h);
        k_maxsum<<<N, 256, 0, stream>>>(attnh, rmax, rsum);
        k_out<<<N / 16, 128, 0, stream>>>(attnh, memn, rmax, rsum, out, h);
    }
}
